// MutualInformationLoss_91130616086866
// MI455X (gfx1250) — compile-verified, matrix operand order checked by reading, not run on hardware
//
#include <hip/hip_runtime.h>
#include <hip/hip_bf16.h>

#define NB 256
#define BM_WORDS 2048   // 256*256 bits / 32

typedef __attribute__((ext_vector_type(16))) _Float16 v16h;
typedef __attribute__((ext_vector_type(8)))  float    v8f;

// ---------------------------------------------------------------- bin helper
__device__ __forceinline__ int bin_of(float vx, float vy) {
    float fx = fminf(fmaxf(vx * 255.0f, 0.0f), 255.0f);
    float fy = fminf(fmaxf(vy * 255.0f, 0.0f), 255.0f);
    return ((int)fx) * NB + (int)fy;   // truncation matches .astype(int32)
}

// ---------------------------------------------------------------- kernel 0: zero the global bitmap
__global__ void k_zero_bitmap(unsigned int* __restrict__ gbm) {
    int t = blockIdx.x * blockDim.x + threadIdx.x;
    if (t < BM_WORDS) gbm[t] = 0u;
}

// ---------------------------------------------------------------- kernel 1: streaming histogram (LDS bitmap)
__global__ __launch_bounds__(256) void k_hist(const float* __restrict__ x,
                                              const float* __restrict__ y,
                                              unsigned int* __restrict__ gbm,
                                              int n) {
    __shared__ unsigned int bm[BM_WORDS];   // 8 KB occupancy bitmap
    const int tid = threadIdx.x;
    for (int w = tid; w < BM_WORDS; w += 256) bm[w] = 0u;
    __syncthreads();

    const int gtid = blockIdx.x * 256 + tid;
    const int nthreads = gridDim.x * 256;
    const int n4 = n >> 2;
    const float4* __restrict__ x4 = (const float4*)x;
    const float4* __restrict__ y4 = (const float4*)y;

    for (int i = gtid; i < n4; i += nthreads) {       // global_load_b128 streaming
        float4 a = x4[i];
        float4 b = y4[i];
        int b0 = bin_of(a.x, b.x);
        int b1 = bin_of(a.y, b.y);
        int b2 = bin_of(a.z, b.z);
        int b3 = bin_of(a.w, b.w);
        atomicOr(&bm[b0 >> 5], 1u << (b0 & 31));      // ds_or_b32, no return
        atomicOr(&bm[b1 >> 5], 1u << (b1 & 31));
        atomicOr(&bm[b2 >> 5], 1u << (b2 & 31));
        atomicOr(&bm[b3 >> 5], 1u << (b3 & 31));
    }
    for (int i = (n4 << 2) + gtid; i < n; i += nthreads) {   // scalar tail
        int b0 = bin_of(x[i], y[i]);
        atomicOr(&bm[b0 >> 5], 1u << (b0 & 31));
    }
    __syncthreads();

    for (int w = tid; w < BM_WORDS; w += 256) {       // merge into global bitmap
        unsigned int v = bm[w];
        if (v) atomicOr(&gbm[w], v);
    }
}

// ---------------------------------------------------------------- kernel 2: bitmap -> float occupancy matrix
__global__ void k_expand(const unsigned int* __restrict__ gbm,
                         float* __restrict__ hist) {
    int w = blockIdx.x * blockDim.x + threadIdx.x;    // one 32-bin word / thread
    if (w >= BM_WORDS) return;
    unsigned int v = gbm[w];
    float4* __restrict__ out = (float4*)(hist + w * 32);
#pragma unroll
    for (int q = 0; q < 8; ++q) {
        float4 f;
        f.x = (v >> (q * 4 + 0)) & 1u ? 1.0f : 0.0f;
        f.y = (v >> (q * 4 + 1)) & 1u ? 1.0f : 0.0f;
        f.z = (v >> (q * 4 + 2)) & 1u ? 1.0f : 0.0f;
        f.w = (v >> (q * 4 + 3)) & 1u ? 1.0f : 0.0f;
        out[q] = f;
    }
}

// ---------------------------------------------------------------- kernel 3: marginals via WMMA, then MI
// Row/col sums of the 256x256 {0,1} matrix are computed as F x ones and
// ones x F using v_wmma_f32_16x16x32_f16 (values 0/1 exact in f16, sums
// <= 256 exact in f32 accumulator). One workgroup = 8 waves; each wave owns
// 2 row-groups and 2 col-groups of 16.
__global__ __launch_bounds__(256) void k_final(const float* __restrict__ hist,
                                               float* __restrict__ out) {
    __shared__ float rowsum[NB];
    __shared__ float colsum[NB];
    __shared__ float red[NB];

    const int tid  = threadIdx.x;
    const int wave = tid >> 5;
    const int lane = tid & 31;
    const int m16  = lane & 15;   // lane % 16
    const int hi   = lane >> 4;   // lane half-group

    v16h ones;
#pragma unroll
    for (int h = 0; h < 16; ++h) ones[h] = (_Float16)1.0f;

    // ---- row sums: D = F_tile(16x32) x ones(32x16), accumulate over K
    for (int gi = 0; gi < 2; ++gi) {
        const int g   = wave * 2 + gi;      // row group 0..15
        const int row = g * 16 + m16;
        v8f acc = {};
        for (int kc = 0; kc < 8; ++kc) {
            v16h a;
#pragma unroll
            for (int h = 0; h < 16; ++h) {
                // ISA 16-bit A 16x32 layout:
                // lanes 0-15 : K = h (h<8)    or h+8  (h>=8)
                // lanes 16-31: K = h+8 (h<8)  or h+16 (h>=8)
                int K = (hi == 0) ? ((h < 8) ? h : h + 8)
                                  : ((h < 8) ? h + 8 : h + 16);
                a[h] = (_Float16)hist[row * NB + kc * 32 + K];
            }
            acc = __builtin_amdgcn_wmma_f32_16x16x32_f16(
                false, a, false, ones, (short)0, acc, false, false);
        }
        // D layout: VGPR r holds M=r (lanes 0-15) and M=r+8 (lanes 16-31),
        // every column identical -> read lane 0 / lane 16.
        if (lane == 0) {
#pragma unroll
            for (int r = 0; r < 8; ++r) rowsum[g * 16 + r] = acc[r];
        }
        if (lane == 16) {
#pragma unroll
            for (int r = 0; r < 8; ++r) rowsum[g * 16 + 8 + r] = acc[r];
        }
    }

    // ---- col sums: D = ones(16x32) x F_tile(32x16), accumulate over K
    for (int gi = 0; gi < 2; ++gi) {
        const int g   = wave * 2 + gi;      // column group 0..15
        const int col = g * 16 + m16;
        v8f acc = {};
        for (int kc = 0; kc < 8; ++kc) {
            v16h b;
#pragma unroll
            for (int h = 0; h < 16; ++h) {
                // 16-bit B 32x16, contiguous K (A is all ones): lanes 0-15 K=0..15, lanes 16-31 K=16..31
                int K = hi ? (16 + h) : h;
                b[h] = (_Float16)hist[(kc * 32 + K) * NB + col];
            }
            acc = __builtin_amdgcn_wmma_f32_16x16x32_f16(
                false, ones, false, b, (short)0, acc, false, false);
        }
        // every row of D identical; lanes 0-15 hold N=lane in acc[0] (M=0)
        if (lane < 16) colsum[g * 16 + lane] = acc[0];
    }
    __syncthreads();

    // ---- S = total occupied bins (sum of rowsums), tree reduce
    red[tid] = rowsum[tid];
    __syncthreads();
    for (int s = 128; s > 0; s >>= 1) {
        if (tid < s) red[tid] += red[tid + s];
        __syncthreads();
    }
    const float S    = red[0];
    __syncthreads();                 // red[] reused below
    const float invS = 1.0f / S;     // h = invS for every occupied bin

    // ---- MI: thread t owns row t
    const float px = rowsum[tid] * invS;
    float local = 0.0f;
    for (int j = 0; j < NB; ++j) {
        if (hist[tid * NB + j] > 0.0f) {
            float py    = colsum[j] * invS;
            float ratio = invS / (px * py + 1e-10f);
            local += invS * logf(ratio);
        }
    }
    red[tid] = local;
    __syncthreads();
    for (int s = 128; s > 0; s >>= 1) {
        if (tid < s) red[tid] += red[tid + s];
        __syncthreads();
    }
    if (tid == 0) out[0] = 1.0f - tanhf(red[0]);
}

// ---------------------------------------------------------------- launch
extern "C" void kernel_launch(void* const* d_in, const int* in_sizes, int n_in,
                              void* d_out, int out_size, void* d_ws, size_t ws_size,
                              hipStream_t stream) {
    const float* x = (const float*)d_in[0];   // I_complementary
    const float* y = (const float*)d_in[1];   // I_target
    float* out = (float*)d_out;
    const int n = in_sizes[0];

    unsigned int* gbm  = (unsigned int*)d_ws;                 // 8 KB bitmap
    float*        hist = (float*)((char*)d_ws + BM_WORDS * 4); // 256 KB occupancy

    k_zero_bitmap<<<(BM_WORDS + 255) / 256, 256, 0, stream>>>(gbm);
    k_hist<<<256, 256, 0, stream>>>(x, y, gbm, n);
    k_expand<<<(BM_WORDS + 255) / 256, 256, 0, stream>>>(gbm, hist);
    k_final<<<1, 256, 0, stream>>>(hist, out);
}
